// DeepseekV4Attention_9637906612573
// MI455X (gfx1250) — matrix operand order repaired and checked by reading, not compiled, not run on hardware
//
#include <hip/hip_runtime.h>
#include <hip/hip_bf16.h>

#define B_ 2
#define S_ 2048
#define D_ 2048
#define H_ 16
#define KVH_ 4
#define HD_ 128
#define LORA_ 512
#define WINDOW_ 1024
#define NEGV (-1000000000.0f)

typedef __attribute__((ext_vector_type(16))) __bf16 bf16x16;
typedef __attribute__((ext_vector_type(8)))  __bf16 bf16x8;
typedef __attribute__((ext_vector_type(8)))  float  f32x8;

static __device__ inline f32x8 wmma_bf16f32(bf16x16 a, bf16x16 b, f32x8 c) {
  // D = A(16x32 bf16) * B(32x16 bf16) + C(16x16 f32)
  return __builtin_amdgcn_wmma_f32_16x16x32_bf16(false, a, false, b, (short)0, c,
                                                 false, false);
}

static __device__ inline f32x8 zero8() {
  f32x8 z;
#pragma unroll
  for (int i = 0; i < 8; ++i) z[i] = 0.0f;
  return z;
}

// Load a 16-element bf16 fragment from two 16-byte chunks.
static __device__ inline bf16x16 load_frag(const __bf16* p0, const __bf16* p1) {
  bf16x8 a = *(const bf16x8*)p0;
  bf16x8 b = *(const bf16x8*)p1;
  bf16x16 r;
#pragma unroll
  for (int i = 0; i < 8; ++i) { r[i] = a[i]; r[i + 8] = b[i]; }
  return r;
}

// ---------------------------------------------------------------- convert
__global__ __launch_bounds__(256) void cvt_f32_to_bf16(const float* __restrict__ s,
                                                       __bf16* __restrict__ d, int n) {
  int i = blockIdx.x * 256 + threadIdx.x;
  if (i < n) d[i] = (__bf16)s[i];
}

// ---------------------------------------------------------------- GEMM
// C[M,N] (f32 or bf16) = A[M,K] * B[K,N], A/B bf16 row-major, f32 accumulate.
// Block: 256 threads = 8 waves; block tile 64(M) x 128(N); K step 32.
// Register-prefetch + ping-pong LDS double buffering: one barrier per K-step.
template <bool OBF>
__global__ __launch_bounds__(256) void gemm_bf16_kernel(const __bf16* __restrict__ A,
                                                        const __bf16* __restrict__ Bm,
                                                        void* __restrict__ Cv,
                                                        int M, int N, int K) {
  __shared__ __align__(16) __bf16 As[2][64 * 32];   // row-major [m][k]
  __shared__ __align__(16) __bf16 Bt[2][128 * 32];  // transposed [n][k]
  const int tid = threadIdx.x;
  const int lane = tid & 31, w = tid >> 5;
  const int m_l = lane & 15, hi = lane >> 4;
  const int m0 = blockIdx.y * 64, n0 = blockIdx.x * 128;
  const int wm = (w & 1) * 32, wn = (w >> 1) * 32;

  // staging thread mapping
  const int arow = tid >> 2, acg = tid & 3;  // A: 64 rows x 32 cols, 16B/thread
  const int bkr = tid >> 3, bng = tid & 7;   // B: 32 rows(K) x 128 cols(N)

  bf16x8 ra, rb0, rb1;
  auto load_regs = [&](int k0) {
    ra = *(const bf16x8*)&A[(size_t)(m0 + arow) * K + k0 + acg * 8];
    const __bf16* src = &Bm[(size_t)(k0 + bkr) * N + n0 + bng * 16];
    rb0 = *(const bf16x8*)src;
    rb1 = *(const bf16x8*)(src + 8);
  };
  auto store_lds = [&](int buf) {
    *(bf16x8*)&As[buf][arow * 32 + acg * 8] = ra;
#pragma unroll
    for (int i = 0; i < 8; ++i) {
      Bt[buf][(bng * 16 + i) * 32 + bkr]     = rb0[i];
      Bt[buf][(bng * 16 + 8 + i) * 32 + bkr] = rb1[i];
    }
  };

  f32x8 acc[2][2];
#pragma unroll
  for (int i = 0; i < 2; ++i)
#pragma unroll
    for (int j = 0; j < 2; ++j) acc[i][j] = zero8();

  load_regs(0);
  store_lds(0);
  __syncthreads();

  for (int k0 = 0; k0 < K; k0 += 32) {
    const int buf = (k0 >> 5) & 1;
    const bool has_next = (k0 + 32) < K;
    if (has_next) load_regs(k0 + 32);  // overlap global loads with WMMAs below

    bf16x16 fa[2], fb[2];
#pragma unroll
    for (int i = 0; i < 2; ++i) {  // A frag: k = 8*hi..+7 and 16+8*hi..+7
      const __bf16* p = &As[buf][(wm + i * 16 + m_l) * 32 + 8 * hi];
      fa[i] = load_frag(p, p + 16);
    }
#pragma unroll
    for (int j = 0; j < 2; ++j) {  // B frag: k = 8*hi..+7 and 16+8*hi..+7 (transposed)
      const __bf16* p = &Bt[buf][(wn + j * 16 + m_l) * 32 + 8 * hi];
      fb[j] = load_frag(p, p + 16);
    }
#pragma unroll
    for (int i = 0; i < 2; ++i)
#pragma unroll
      for (int j = 0; j < 2; ++j) acc[i][j] = wmma_bf16f32(fa[i], fb[j], acc[i][j]);

    if (has_next) store_lds(buf ^ 1);
    __syncthreads();
  }

  // epilogue: C layout -> row = v + 8*hi, col = lane%16
#pragma unroll
  for (int i = 0; i < 2; ++i)
#pragma unroll
    for (int j = 0; j < 2; ++j)
#pragma unroll
      for (int v = 0; v < 8; ++v) {
        int row = m0 + wm + i * 16 + v + 8 * hi;
        int col = n0 + wn + j * 16 + m_l;
        float val = acc[i][j][v];
        if (OBF)
          ((__bf16*)Cv)[(size_t)row * N + col] = (__bf16)val;
        else
          ((float*)Cv)[(size_t)row * N + col] = val;
      }
}

// ---------------------------------------------------------------- partial RoPE (in place)
// x: [B,S,nheads,128] bf16. One thread per (token, head, d) with d in [0,32).
__global__ __launch_bounds__(256) void rope_kernel(__bf16* __restrict__ x,
                                                   const int* __restrict__ pos_ids,
                                                   int nheads) {
  int t = blockIdx.x * 256 + threadIdx.x;
  int total = B_ * S_ * nheads * 32;
  if (t >= total) return;
  int d = t & 31;
  int h = (t >> 5) % nheads;
  int tok = t / (32 * nheads);  // = b*S + s
  float pos = (float)pos_ids[tok];
  float inv_freq = __powf(10000.0f, -(float)d / 32.0f);
  float f = pos * inv_freq;
  float sn, cs;
  __sincosf(f, &sn, &cs);
  size_t base = (size_t)tok * nheads * HD_ + (size_t)h * HD_;
  float x1 = (float)x[base + d];
  float x2 = (float)x[base + 32 + d];
  x[base + d]      = (__bf16)(x1 * cs - x2 * sn);
  x[base + 32 + d] = (__bf16)(x2 * cs + x1 * sn);
}

// ---------------------------------------------------------------- flash attention
// q: [B,S,H,128], k/v: [B,S,KVH,128] bf16 (rope applied); o: [B,S,H,128] bf16.
// Grid (S/128, H, B); 8 waves per block, each wave owns 16 query rows.
// K/V tiles double-buffered in LDS with register prefetch.
__global__ __launch_bounds__(256) void attn_kernel(const __bf16* __restrict__ q,
                                                   const __bf16* __restrict__ kk,
                                                   const __bf16* __restrict__ vv,
                                                   __bf16* __restrict__ o) {
  __shared__ __align__(16) __bf16 ldsK[2][32 * 128];   // [key][hd]
  __shared__ __align__(16) __bf16 ldsVt[2][128 * 32];  // [hd][key] (transposed)
  __shared__ __align__(16) __bf16 ldsP[8 * 16 * 32];   // per-wave P tile [row][key]
  const int tid = threadIdx.x, lane = tid & 31, w = tid >> 5;
  const int m_l = lane & 15, hi = lane >> 4;
  const int qt = blockIdx.x, h = blockIdx.y, b = blockIdx.z;
  const int kvh = h >> 2;  // GQA: rep = H/KVH = 4
  const int q0b = qt * 128;
  const int qbase = q0b + w * 16;
  const float scale = 0.08838834764831845f;  // 1/sqrt(128)

  const size_t kvld = (size_t)(KVH_ * HD_);
  const __bf16* kbase_p = kk + (size_t)b * S_ * kvld + kvh * HD_;
  const __bf16* vbase_p = vv + (size_t)b * S_ * kvld + kvh * HD_;

  // staging thread mapping
  const int kkey0 = tid >> 4, kcg0 = tid & 15;         // K chunk 0 (chunks 0..255)
  const int kkey1 = (tid + 256) >> 4, kcg1 = tid & 15; // K chunk 1 (chunks 256..511)
  const int vkey = tid >> 3, vcg = tid & 7;            // V: 16 elems/thread

  bf16x8 rk0, rk1, rv0, rv1;
  auto load_kv_regs = [&](int kb) {
    rk0 = *(const bf16x8*)&kbase_p[(size_t)(kb + kkey0) * kvld + kcg0 * 8];
    rk1 = *(const bf16x8*)&kbase_p[(size_t)(kb + kkey1) * kvld + kcg1 * 8];
    const __bf16* src = &vbase_p[(size_t)(kb + vkey) * kvld + vcg * 16];
    rv0 = *(const bf16x8*)src;
    rv1 = *(const bf16x8*)(src + 8);
  };
  auto store_kv_lds = [&](int buf) {
    *(bf16x8*)&ldsK[buf][kkey0 * 128 + kcg0 * 8] = rk0;
    *(bf16x8*)&ldsK[buf][kkey1 * 128 + kcg1 * 8] = rk1;
#pragma unroll
    for (int i = 0; i < 8; ++i) {
      ldsVt[buf][(vcg * 16 + i) * 32 + vkey]     = rv0[i];
      ldsVt[buf][(vcg * 16 + 8 + i) * 32 + vkey] = rv1[i];
    }
  };

  // Q fragments: 16 rows x 128 (4 K-chunks of 32)
  bf16x16 QA[4];
  {
    const __bf16* qrow = q + (size_t)(b * S_ + qbase + m_l) * (H_ * HD_) + h * HD_;
#pragma unroll
    for (int c = 0; c < 4; ++c) {
      const __bf16* p = qrow + c * 32 + 8 * hi;
      QA[c] = load_frag(p, p + 16);
    }
  }

  f32x8 O[8];
#pragma unroll
  for (int i = 0; i < 8; ++i) O[i] = zero8();
  float mi[8], li[8];
#pragma unroll
  for (int v = 0; v < 8; ++v) { mi[v] = -1.0e30f; li[v] = 0.0f; }

  int jlo = q0b - (WINDOW_ - 1);
  if (jlo < 0) jlo = 0;
  jlo &= ~31;  // block-uniform loop bounds -> __syncthreads is legal inside

  load_kv_regs(jlo);
  store_kv_lds(0);
  __syncthreads();

  for (int kb = jlo; kb < q0b + 128; kb += 32) {
    const int buf = ((kb - jlo) >> 5) & 1;
    const bool has_next = (kb + 32) < q0b + 128;
    if (has_next) load_kv_regs(kb + 32);  // overlap with score/softmax work
    if (kb + 64 < q0b + 128) {            // warm L2 two tiles ahead (global_prefetch_b8)
      __builtin_prefetch(&kbase_p[(size_t)(kb + 64 + kkey0) * kvld], 0, 1);
      __builtin_prefetch(&vbase_p[(size_t)(kb + 64 + vkey) * kvld], 0, 1);
    }

    // scores: S[16 q][32 keys] = Q(16x128) * K^T(128x32)
    f32x8 s0 = zero8(), s1 = zero8();
#pragma unroll
    for (int c = 0; c < 4; ++c) {
      const __bf16* p0 = &ldsK[buf][m_l * 128 + c * 32 + 8 * hi];
      const __bf16* p1 = &ldsK[buf][(16 + m_l) * 128 + c * 32 + 8 * hi];
      bf16x16 kf0 = load_frag(p0, p0 + 16);
      bf16x16 kf1 = load_frag(p1, p1 + 16);
      s0 = wmma_bf16f32(QA[c], kf0, s0);
      s1 = wmma_bf16f32(QA[c], kf1, s1);
    }

    // online softmax per row (row across 16 lanes of a half-wave)
    float alphav[8];
#pragma unroll
    for (int v = 0; v < 8; ++v) {
      int irow = qbase + v + 8 * hi;
      int j0 = kb + m_l, j1 = kb + 16 + m_l;
      bool ok0 = (j0 <= irow) && (irow - j0 < WINDOW_);
      bool ok1 = (j1 <= irow) && (irow - j1 < WINDOW_);
      float a0 = ok0 ? s0[v] * scale : NEGV;
      float a1 = ok1 ? s1[v] * scale : NEGV;
      float mx = fmaxf(a0, a1);
#pragma unroll
      for (int off = 1; off < 16; off <<= 1) mx = fmaxf(mx, __shfl_xor(mx, off, 32));
      float mnew = fmaxf(mi[v], mx);
      float alpha = __expf(mi[v] - mnew);
      float p0v = ok0 ? __expf(a0 - mnew) : 0.0f;
      float p1v = ok1 ? __expf(a1 - mnew) : 0.0f;
      float ps = p0v + p1v;
#pragma unroll
      for (int off = 1; off < 16; off <<= 1) ps += __shfl_xor(ps, off, 32);
      li[v] = li[v] * alpha + ps;
      mi[v] = mnew;
      alphav[v] = alpha;
      int prow = v + 8 * hi;  // C-layout row owned by this lane
      ldsP[w * 512 + prow * 32 + m_l]      = (__bf16)p0v;
      ldsP[w * 512 + prow * 32 + 16 + m_l] = (__bf16)p1v;
    }
    __syncthreads();  // P visible (C-layout -> A-layout via LDS)

#pragma unroll
    for (int hc = 0; hc < 8; ++hc)
#pragma unroll
      for (int v = 0; v < 8; ++v) O[hc][v] *= alphav[v];

    bf16x16 PA;
    {
      const __bf16* p = &ldsP[w * 512 + m_l * 32 + 8 * hi];
      PA = load_frag(p, p + 16);
    }
#pragma unroll
    for (int hc = 0; hc < 8; ++hc) {  // O(16x128) += P(16x32) * V(32x128)
      const __bf16* p = &ldsVt[buf][(hc * 16 + m_l) * 32 + 8 * hi];
      bf16x16 vf = load_frag(p, p + 16);
      O[hc] = wmma_bf16f32(PA, vf, O[hc]);
    }

    if (has_next) store_kv_lds(buf ^ 1);
    __syncthreads();
  }

  // epilogue: normalize and store
#pragma unroll
  for (int v = 0; v < 8; ++v) {
    float inv = 1.0f / li[v];
    int row = qbase + v + 8 * hi;
    __bf16* orow = o + (size_t)(b * S_ + row) * (H_ * HD_) + h * HD_;
#pragma unroll
    for (int hc = 0; hc < 8; ++hc) orow[hc * 16 + m_l] = (__bf16)(O[hc][v] * inv);
  }
}

// ---------------------------------------------------------------- driver
extern "C" void kernel_launch(void* const* d_in, const int* in_sizes, int n_in,
                              void* d_out, int out_size, void* d_ws, size_t ws_size,
                              hipStream_t stream) {
  (void)in_sizes; (void)n_in; (void)out_size; (void)ws_size;
  const float* hidden = (const float*)d_in[0];
  const int*   pos    = (const int*)d_in[1];
  const float* Wqa    = (const float*)d_in[2];
  const float* Wqb    = (const float*)d_in[3];
  const float* Wk     = (const float*)d_in[4];
  const float* Wv     = (const float*)d_in[5];
  const float* Wo     = (const float*)d_in[6];
  float* out = (float*)d_out;

  char* p = (char*)d_ws;
  auto alloc = [&](size_t elems) {
    __bf16* r = (__bf16*)p;
    p += ((elems * sizeof(__bf16)) + 255) & ~(size_t)255;
    return r;
  };
  const size_t M = (size_t)B_ * S_;  // 4096
  __bf16* hid_bf = alloc(M * D_);
  __bf16* Wqa_bf = alloc((size_t)D_ * LORA_);
  __bf16* Wqb_bf = alloc((size_t)LORA_ * H_ * HD_);
  __bf16* Wk_bf  = alloc((size_t)D_ * KVH_ * HD_);
  __bf16* Wv_bf  = alloc((size_t)D_ * KVH_ * HD_);
  __bf16* Wo_bf  = alloc((size_t)H_ * HD_ * D_);
  __bf16* qa_bf  = alloc(M * LORA_);
  __bf16* q_bf   = alloc(M * H_ * HD_);
  __bf16* k_bf   = alloc(M * KVH_ * HD_);
  __bf16* v_bf   = alloc(M * KVH_ * HD_);
  __bf16* at_bf  = alloc(M * H_ * HD_);

  auto cvt = [&](const float* s, __bf16* d, size_t n) {
    cvt_f32_to_bf16<<<dim3((unsigned)((n + 255) / 256)), dim3(256), 0, stream>>>(
        s, d, (int)n);
  };
  cvt(hidden, hid_bf, M * D_);
  cvt(Wqa, Wqa_bf, (size_t)D_ * LORA_);
  cvt(Wqb, Wqb_bf, (size_t)LORA_ * H_ * HD_);
  cvt(Wk, Wk_bf, (size_t)D_ * KVH_ * HD_);
  cvt(Wv, Wv_bf, (size_t)D_ * KVH_ * HD_);
  cvt(Wo, Wo_bf, (size_t)H_ * HD_ * D_);

  // qa = hidden @ Wqa   [4096 x 512]
  gemm_bf16_kernel<true><<<dim3(LORA_ / 128, (unsigned)(M / 64)), 256, 0, stream>>>(
      hid_bf, Wqa_bf, qa_bf, (int)M, LORA_, D_);
  // q = qa @ Wqb        [4096 x 2048]
  gemm_bf16_kernel<true><<<dim3((H_ * HD_) / 128, (unsigned)(M / 64)), 256, 0, stream>>>(
      qa_bf, Wqb_bf, q_bf, (int)M, H_ * HD_, LORA_);
  // k = hidden @ Wk     [4096 x 512]
  gemm_bf16_kernel<true><<<dim3((KVH_ * HD_) / 128, (unsigned)(M / 64)), 256, 0, stream>>>(
      hid_bf, Wk_bf, k_bf, (int)M, KVH_ * HD_, D_);
  // v = hidden @ Wv     [4096 x 512]
  gemm_bf16_kernel<true><<<dim3((KVH_ * HD_) / 128, (unsigned)(M / 64)), 256, 0, stream>>>(
      hid_bf, Wv_bf, v_bf, (int)M, KVH_ * HD_, D_);

  // partial RoPE in place on q and k
  rope_kernel<<<dim3((B_ * S_ * H_ * 32 + 255) / 256), 256, 0, stream>>>(q_bf, pos, H_);
  rope_kernel<<<dim3((B_ * S_ * KVH_ * 32 + 255) / 256), 256, 0, stream>>>(k_bf, pos, KVH_);

  // sliding-window flash attention
  attn_kernel<<<dim3(S_ / 128, H_, B_), 256, 0, stream>>>(q_bf, k_bf, v_bf, at_bf);

  // out = attn @ Wo     [4096 x 2048] f32
  gemm_bf16_kernel<false><<<dim3(D_ / 128, (unsigned)(M / 64)), 256, 0, stream>>>(
      at_bf, Wo_bf, (void*)out, (int)M, D_, H_ * HD_);
}
